// DCCNN_50397146251352
// MI455X (gfx1250) — matrix operand order repaired and checked by reading, not compiled, not run on hardware
//
#include <hip/hip_runtime.h>
#include <hip/hip_bf16.h>
#include <math.h>

typedef __attribute__((ext_vector_type(16))) _Float16 v16h;
typedef __attribute__((ext_vector_type(8)))  _Float16 h8;
typedef __attribute__((ext_vector_type(4)))  _Float16 h4;
typedef __attribute__((ext_vector_type(8)))  float    v8f;

#define NODES   204800
#define NEDGES  1638400
#define NGRAPH  4096
#define NPG     50
#define FEAT    192
#define KTOP    10
#define RPB     64        // GEMM rows per block

// ---------------- degree / dinv ----------------
__global__ void k_deg_init(float* __restrict__ deg) {
  int n = blockIdx.x * blockDim.x + threadIdx.x;
  if (n < NODES) deg[n] = 1.0f;                       // +1 self loop
}
__global__ void k_deg_count(const int* __restrict__ col, float* __restrict__ deg) {
  int e = blockIdx.x * blockDim.x + threadIdx.x;
  if (e < NEDGES) atomicAdd(&deg[col[e]], 1.0f);
}
__global__ void k_dinv(const float* __restrict__ deg, float* __restrict__ dinv) {
  int n = blockIdx.x * blockDim.x + threadIdx.x;
  if (n < NODES) dinv[n] = rsqrtf(deg[n]);
}

// ---------------- W pre-swizzle: row-major f32 [K,64] -> WMMA B-fragment order f16 ----
// dst index for (k,n): chunk c=k/32, half h=(k/16)&1, q=k&15 -> ((c*64+n)*2+h)*16+q
// so each lane's 16 B values (col n, K = 32c + 16h + q) are 32 contiguous bytes.
__global__ void k_prepw(const float* __restrict__ W, _Float16* __restrict__ wsw, int K) {
  int i = blockIdx.x * blockDim.x + threadIdx.x;
  if (i >= K * 64) return;
  int k = i / 64, n = i - k * 64;
  int c = k >> 5, h = (k >> 4) & 1, q = k & 15;
  wsw[((c * 64 + n) * 2 + h) * 16 + q] = (_Float16)W[i];
}

// ---------------- WMMA GEMM: H = X @ W ; Agg = H*dinv^2 + b ----------------
// block = 128 threads (4 waves). Block computes RPB=64 rows x 64 cols.
// Each wave owns a 16-col slice and 4 row-tiles sharing one B fragment per K-chunk.
__launch_bounds__(128)
__global__ void k_gemm_wmma(const float* __restrict__ X, int xstride, int K,
                            const _Float16* __restrict__ Wsw, const float* __restrict__ bias,
                            const float* __restrict__ dinv,
                            float* __restrict__ H, float* __restrict__ Agg) {
  __shared__ _Float16 Wh[128 * 64];     // swizzled fragments, up to 16KB
  __shared__ _Float16 Xh[RPB * 128];    // row-major pitch K, up to 16KB
  const int tid  = threadIdx.x;
  const int row0 = blockIdx.x * RPB;

  // stage swizzled W: raw uint4 copy (no conversion, no swizzle at block level)
  {
    const uint4* src = (const uint4*)Wsw;
    uint4* dst = (uint4*)Wh;
    const int nw = (K * 64 * 2) / 16;               // 1024 (K=128) or 512 (K=64)
    for (int i = tid; i < nw; i += 128) dst[i] = src[i];
  }
  // stage X tile with f32->f16 conversion, vectorized by 4
  {
    const int nx = (RPB * K) / 4;
    for (int i = tid; i < nx; i += 128) {
      int e = i * 4;
      int r = e / K, k = e - r * K;
      float4 v = *(const float4*)&X[(long long)(row0 + r) * xstride + k];
      h4 hv = {(_Float16)v.x, (_Float16)v.y, (_Float16)v.z, (_Float16)v.w};
      *(h4*)&Xh[r * K + k] = hv;
    }
  }
  __syncthreads();

  const int lane = tid & 31;
  const int wave = tid >> 5;
  const int half = lane >> 4;
  const int m    = lane & 15;
  const int col  = wave * 16 + m;

  v8f acc0 = {0.f,0.f,0.f,0.f,0.f,0.f,0.f,0.f};
  v8f acc1 = acc0, acc2 = acc0, acc3 = acc0;

  for (int kk = 0, c = 0; kk < K; kk += 32, ++c) {
    // B fragment: 32 contiguous bytes -> 2x ds_load_b128
    v16h b = *(const v16h*)&Wh[((c * 64 + col) * 2 + half) * 16];
#pragma unroll
    for (int t = 0; t < 4; ++t) {
      const _Float16* xr = &Xh[(t * 16 + m) * K + kk + half * 16];
      h8 lo = *(const h8*)xr;            // K = kk + half*16 + [0..7]
      h8 hi = *(const h8*)(xr + 8);      // K = kk + half*16 + 8 + [0..7]
      v16h a;
#pragma unroll
      for (int i = 0; i < 8; ++i) { a[i] = lo[i]; a[8 + i] = hi[i]; }
      v8f* pa = (t == 0) ? &acc0 : (t == 1) ? &acc1 : (t == 2) ? &acc2 : &acc3;
      *pa = __builtin_amdgcn_wmma_f32_16x16x32_f16(false, a, false, b,
                                                   (short)0, *pa, false, false);
    }
  }

  const float bcol = bias[col];
#pragma unroll
  for (int t = 0; t < 4; ++t) {
    v8f acc = (t == 0) ? acc0 : (t == 1) ? acc1 : (t == 2) ? acc2 : acc3;
#pragma unroll
    for (int r = 0; r < 8; ++r) {
      int node = row0 + t * 16 + r + 8 * half;      // M = r + 8*half within tile t
      float v  = acc[r];
      float di = dinv[node];
      long long idx = (long long)node * 64 + col;
      H[idx]   = v;
      Agg[idx] = v * di * di + bcol;                // self-loop + bias
    }
  }
}

// ---------------- edge scatter: agg[col] += h[row]*dinv[row]*dinv[col] ----------------
__global__ void k_scatter(const int* __restrict__ row, const int* __restrict__ col,
                          const float* __restrict__ dinv, const float* __restrict__ h,
                          float* __restrict__ agg) {
  long long t = (long long)blockIdx.x * blockDim.x + threadIdx.x;
  long long e = t >> 5;
  int lane = (int)(t & 31);
  if (e >= NEDGES) return;
  int r = row[e], c = col[e];
  float coef = dinv[r] * dinv[c];
  long long rb = (long long)r * 64, cb = (long long)c * 64;
  atomicAdd(&agg[cb + lane],      h[rb + lane]      * coef);
  atomicAdd(&agg[cb + 32 + lane], h[rb + 32 + lane] * coef);
}

// ---------------- tanh into feat slice ----------------
__global__ void k_tanh(const float* __restrict__ agg, float* __restrict__ feat, int off) {
  long long t = (long long)blockIdx.x * blockDim.x + threadIdx.x;
  if (t >= (long long)NODES * 64) return;
  int n = (int)(t >> 6), c = (int)(t & 63);
  feat[(long long)n * FEAT + off + c] = tanhf(agg[t]);
}

// ---------------- per-node ascending sort of 192 features (bitonic over 256) ----------------
__launch_bounds__(256)
__global__ void k_sort(float* __restrict__ feat) {
  __shared__ float s[256];
  int n = blockIdx.x, t = threadIdx.x;
  s[t] = (t < FEAT) ? feat[(long long)n * FEAT + t] : 3.0e38f;
  __syncthreads();
  for (int k = 2; k <= 256; k <<= 1) {
    for (int j = k >> 1; j > 0; j >>= 1) {
      int ixj = t ^ j;
      if (ixj > t) {
        bool up = ((t & k) == 0);
        float a = s[t], b = s[ixj];
        if ((a > b) == up) { s[t] = b; s[ixj] = a; }
      }
      __syncthreads();
    }
  }
  if (t < FEAT) feat[(long long)n * FEAT + t] = s[t];
}

// ---------------- per-graph stable top-K by last feature, gather pooled ----------------
__launch_bounds__(64)
__global__ void k_topk(const float* __restrict__ feat, float* __restrict__ pooled) {
  __shared__ float key[NPG];
  __shared__ int   sel[KTOP];
  int g = blockIdx.x, t = threadIdx.x;
  if (t < NPG) key[t] = feat[(long long)(g * NPG + t) * FEAT + (FEAT - 1)];
  __syncthreads();
  if (t == 0) {
    for (int r = 0; r < KTOP; ++r) {
      int best = 0; float bk = key[0];
      for (int i = 1; i < NPG; ++i)
        if (key[i] > bk) { bk = key[i]; best = i; }    // strict > = stable tie-break
      sel[r] = best; key[best] = -3.0e38f;
    }
  }
  __syncthreads();
  for (int r = 0; r < KTOP; ++r) {
    long long src = (long long)(g * NPG + sel[r]) * FEAT;
    long long dst = (long long)(g * KTOP + r) * FEAT;
    for (int c = t; c < FEAT; c += 64) pooled[dst + c] = feat[src + c];
  }
}

// ---------------- conv1 (C=10, L=192, k=4 s=4) + relu + maxpool4 -> p1[B,32,12] ----------------
__launch_bounds__(384)
__global__ void k_conv1(const float* __restrict__ pooled, const float* __restrict__ Wc1,
                        const float* __restrict__ bc1, float* __restrict__ p1) {
  __shared__ float ps[KTOP * FEAT];     // 1920
  __shared__ float ws[32 * KTOP * 4];   // 1280
  int b = blockIdx.x, tid = threadIdx.x;
  for (int i = tid; i < KTOP * FEAT; i += 384) ps[i] = pooled[(long long)b * (KTOP * FEAT) + i];
  for (int i = tid; i < 1280; i += 384) ws[i] = Wc1[i];
  __syncthreads();
  int o = tid / 12, u = tid % 12;       // 32 x 12 = 384
  float mx = -3.0e38f;
  for (int t = 4 * u; t < 4 * u + 4; ++t) {
    float s = bc1[o];
    for (int i = 0; i < KTOP; ++i) {
#pragma unroll
      for (int k = 0; k < 4; ++k)
        s += ps[i * FEAT + 4 * t + k] * ws[o * 40 + i * 4 + k];
    }
    mx = fmaxf(mx, fmaxf(s, 0.0f));
  }
  p1[(long long)b * 384 + o * 12 + u] = mx;
}

// ---------------- conv2 (C=32, L=12, k=3 s=3) + relu + maxpool4 + FC ----------------
__launch_bounds__(256)
__global__ void k_conv2(const float* __restrict__ p1, const float* __restrict__ Wc2,
                        const float* __restrict__ bc2, const float* __restrict__ Wf,
                        const float* __restrict__ bf, float* __restrict__ out_c,
                        float* __restrict__ out_xf) {
  __shared__ float ps[384];
  __shared__ float ws[64 * 32 * 3];     // 6144
  __shared__ float xfs[64];
  int b = blockIdx.x, tid = threadIdx.x;
  for (int i = tid; i < 384;  i += 256) ps[i] = p1[(long long)b * 384 + i];
  for (int i = tid; i < 6144; i += 256) ws[i] = Wc2[i];
  __syncthreads();
  if (tid < 64) {
    int o = tid;
    float mx = -3.0e38f;
    for (int t = 0; t < 4; ++t) {
      float s = bc2[o];
      for (int i = 0; i < 32; ++i) {
#pragma unroll
        for (int k = 0; k < 3; ++k)
          s += ps[i * 12 + 3 * t + k] * ws[o * 96 + i * 3 + k];
      }
      mx = fmaxf(mx, fmaxf(s, 0.0f));
    }
    xfs[o] = mx;
    out_xf[(long long)b * 64 + o] = mx;
  }
  __syncthreads();
  if (tid < 10) {
    float s = bf[tid];
    for (int o = 0; o < 64; ++o)
      s += fmaxf(xfs[o], 0.0f) * Wf[o * 10 + tid];
    out_c[(long long)b * 10 + tid] = s;
  }
}

extern "C" void kernel_launch(void* const* d_in, const int* in_sizes, int n_in,
                              void* d_out, int out_size, void* d_ws, size_t ws_size,
                              hipStream_t stream) {
  (void)in_sizes; (void)n_in; (void)out_size; (void)ws_size;
  const float* x    = (const float*)d_in[0];
  const int*   ei   = (const int*)d_in[1];
  const int*   rowp = ei;
  const int*   colp = ei + NEDGES;
  // d_in[2] = batch: structure is known (graph g owns nodes [g*50,(g+1)*50))
  const float* W1  = (const float*)d_in[3];  const float* b1  = (const float*)d_in[4];
  const float* W2  = (const float*)d_in[5];  const float* b2  = (const float*)d_in[6];
  const float* W3  = (const float*)d_in[7];  const float* b3  = (const float*)d_in[8];
  const float* Wc1 = (const float*)d_in[9];  const float* bc1 = (const float*)d_in[10];
  const float* Wc2 = (const float*)d_in[11]; const float* bc2 = (const float*)d_in[12];
  const float* Wf  = (const float*)d_in[13]; const float* bf  = (const float*)d_in[14];

  float* ws   = (float*)d_ws;
  float* deg  = ws;                               // N
  float* dinv = deg  + NODES;                     // N
  float* h    = dinv + NODES;                     // N*64
  float* agg  = h    + (long long)NODES * 64;     // N*64
  float* feat = agg  + (long long)NODES * 64;     // N*192
  _Float16* wsw = (_Float16*)(feat + (long long)NODES * FEAT);  // 128*64 f16
  float* pooled = h;                              // reuse after GCN stage (7.9M <= 13.1M)
  float* p1     = agg;                            // reuse (1.6M <= 13.1M)

  float* out_c  = (float*)d_out;                  // [B,10]
  float* out_xf = out_c + NGRAPH * 10;            // [B,64]

  k_deg_init <<<(NODES  + 255) / 256, 256, 0, stream>>>(deg);
  k_deg_count<<<(NEDGES + 255) / 256, 256, 0, stream>>>(colp, deg);
  k_dinv     <<<(NODES  + 255) / 256, 256, 0, stream>>>(deg, dinv);

  const float* Xs[3]      = {x, feat, feat + 64};
  const int    strides[3] = {128, 192, 192};
  const int    Ks[3]      = {128, 64, 64};
  const float* Wl[3]      = {W1, W2, W3};
  const float* bl[3]      = {b1, b2, b3};

  for (int l = 0; l < 3; ++l) {
    const int K = Ks[l];
    k_prepw    <<<(K * 64 + 255) / 256, 256, 0, stream>>>(Wl[l], wsw, K);
    k_gemm_wmma<<<NODES / RPB, 128, 0, stream>>>(Xs[l], strides[l], K,
                                                 wsw, bl[l], dinv, h, agg);
    k_scatter  <<<(int)(((long long)NEDGES * 32) / 256), 256, 0, stream>>>(rowp, colp, dinv, h, agg);
    k_tanh     <<<(int)(((long long)NODES * 64) / 256), 256, 0, stream>>>(agg, feat, l * 64);
  }

  k_sort <<<NODES, 256, 0, stream>>>(feat);
  k_topk <<<NGRAPH, 64, 0, stream>>>(feat, pooled);
  k_conv1<<<NGRAPH, 384, 0, stream>>>(pooled, Wc1, bc1, p1);
  k_conv2<<<NGRAPH, 256, 0, stream>>>(p1, Wc2, bc2, Wf, bf, out_c, out_xf);
}
